// GCK3x3Layer_24412594110720
// MI455X (gfx1250) — matrix operand order repaired and checked by reading, not compiled, not run on hardware
//
#include <hip/hip_runtime.h>
#include <hip/hip_bf16.h>
#include <stdint.h>

typedef __attribute__((ext_vector_type(16))) __bf16 v16bf;
typedef __attribute__((ext_vector_type(8)))  float  v8f;
typedef __attribute__((ext_vector_type(4)))  unsigned int v4u;
typedef __attribute__((ext_vector_type(8)))  int  v8i;
typedef __attribute__((ext_vector_type(4)))  int  v4i;

namespace gck {
constexpr int C    = 64;
constexpr int O    = 64;
constexpr int Hin  = 514;
constexpr int Win  = 514;
constexpr int Ho   = 512;
constexpr int Wo   = 512;
constexpr int TILE_P = 128;          // pixels per block (one output row)
constexpr int KTOT   = 192;          // k = r*64 + c, per s-shift (3 shifts -> K=576)
constexpr int KPAD   = 208;          // LDS col stride (208*2 = 416 B = 13*32B: odd multiple
                                     // of 32B -> B-frag reads spread across all bank groups)
constexpr int NCOL   = TILE_P + 2;   // 130 columns staged
constexpr int NFRAG  = 3 * 6 * 4;    // s * kchunk * mtile A fragments
constexpr int APACK_ELTS = NFRAG * 32 * 16;          // 36864 bf16
constexpr int APACK_DW   = APACK_ELTS * 2 / 4;       // 18432 dwords = 0x4800
}

// ---------------------------------------------------------------------------
// Kernel 1: fold linCombs (O x C*9) through the +/-1 GCK basis into effective
// 3x3 kernels, emitted directly in WMMA A-fragment layout (bf16):
//   apack[ ((s*6 + kc)*4 + m)*32*16 + lane*16 + e ]
// A 16x32 bf16 k order used here (not the ISA table order; B uses the same): lane L -> M = L%16 ; element e (vgpr v=e/2, half p=e%2):
//   Klocal = 2v + p + 16*(L/16)
// Global k = kc*32 + Klocal ; r = k/64 ; c = k%64 ; o = m*16 + M.
// ---------------------------------------------------------------------------
__global__ __launch_bounds__(256) void gck_fold_kernels(
    const float* __restrict__ linCombs, __bf16* __restrict__ apack) {
  int idx = blockIdx.x * 256 + threadIdx.x;
  if (idx >= gck::APACK_ELTS) return;

  int e    = idx & 15;
  int lane = (idx >> 4) & 31;
  int m    = (idx >> 9) & 3;
  int t    = idx >> 11;          // 0..17
  int kc   = t % 6;
  int s    = t / 6;

  int hi = lane >> 4;
  int M  = lane & 15;
  int v  = e >> 1;
  int p  = e & 1;
  int klocal = 2 * v + p + 16 * hi;
  int k = kc * 32 + klocal;      // 0..191
  int r = k >> 6;                // 0..2
  int c = k & 63;
  int o = m * 16 + M;

  const float Vm[3][3] = {{1.f, 1.f, 1.f}, {1.f, -1.f, 1.f}, {1.f, 1.f, -1.f}};
  float w = 0.f;
#pragma unroll
  for (int i = 0; i < 3; ++i)
#pragma unroll
    for (int j = 0; j < 3; ++j)
      w += linCombs[o * (gck::C * 9) + c * 9 + 3 * i + j] * Vm[i][r] * Vm[j][s];

  apack[idx] = (__bf16)w;
}

// ---------------------------------------------------------------------------
// TDM: broadcast the whole packed-A matrix (73728 B, already bf16, already in
// fragment layout) into LDS as one 1-D tensor tile of 18432 dwords.
// D# group0: count=1, lds_addr, 57-bit global addr, type=2.
// D# group1: data_size=4B, tensor_dim0 = tile_dim0 = 18432, dims1.. = 1/0.
// amdgpu-toolchain 6-arg builtin form: (v4u, v8i, v4i, v4i, v8i, i32 cpol).
// ---------------------------------------------------------------------------
__device__ __forceinline__ void tdm_load_apack(const __bf16* gsrc, __bf16* ldst) {
  uint64_t ga   = (uint64_t)(uintptr_t)gsrc;
  uint32_t galo = (uint32_t)ga;
  uint32_t gahi = (uint32_t)(ga >> 32);
  uint32_t ldsa = (uint32_t)(uintptr_t)ldst;     // LDS byte offset (low 32 bits)
  v4u g0 = { 1u,                                 // count=1 (valid descriptor)
             ldsa,                               // lds_addr
             galo,                               // global_addr[31:0]
             (gahi & 0x01FFFFFFu) | (2u << 30) };// global_addr[56:32] | type=2
  v8i g1 = { (int)0x00020000,                    // data_size=2 (4B), mask/flags=0
             (int)(gck::APACK_DW << 16),         // tensor_dim0[15:0] in [31:16]
             (int)0x00010000,                    // tensor_dim0[31:16]=0; tensor_dim1=1
             (int)(gck::APACK_DW << 16),         // tensor_dim1 hi=0; tile_dim0=18432
             (int)0x00000001,                    // tile_dim1=1, tile_dim2=0
             (int)gck::APACK_DW,                 // tensor_dim0_stride[31:0]
             0, 0 };
  v4i g2 = {0, 0, 0, 0};
  v4i g3 = {0, 0, 0, 0};
  v8i g4 = {0, 0, 0, 0, 0, 0, 0, 0};
  __builtin_amdgcn_tensor_load_to_lds(g0, g1, g2, g3, g4, 0);
}

// ---------------------------------------------------------------------------
// Kernel 2: implicit-GEMM 3x3 conv. Block = 256 threads = 8 waves.
// Block tile: one output row y, 128 pixels, all 64 outputs.
// Wave w: pixels [w*16, w*16+16), 4 M-tiles (all O), 4x v8f accumulators.
// x tile staged transposed+bf16 in LDS as [col 0..129][k = r*64+c] so each
// B fragment (32x16 bf16: lane L -> N=L%16, K = e + 16*(L/16)) is one
// contiguous 32B ds read. Packed A lives in LDS via TDM (wave 0 issues,
// S_WAIT_TENSORCNT + barrier), each fragment = 1024 B contiguous per wave.
// ---------------------------------------------------------------------------
__global__ __launch_bounds__(256) void gck_conv_wmma(
    const float* __restrict__ x, const __bf16* __restrict__ apack,
    float* __restrict__ out) {
  __shared__ __align__(128) __bf16 xpanel[gck::NCOL * gck::KPAD];   // 54080 B
  __shared__ __align__(128) __bf16 apanel[gck::APACK_ELTS];          // 73728 B

  const int tid = threadIdx.x;
  const int y   = blockIdx.x >> 2;            // 512 rows
  const int x0  = (blockIdx.x & 3) * gck::TILE_P;

  // ---- kick off async TDM broadcast of packed A into LDS (wave 0 only) ----
  if (tid < 32) tdm_load_apack(apack, apanel);

  // ---- stage x tile into LDS, transposed + converted to bf16 -------------
  // Row bases are always 8B-aligned ((c*514+y+r)*514 is even), so use float2.
  for (int idx = tid; idx < gck::KTOT * (gck::NCOL / 2); idx += 256) {
    int k   = idx / (gck::NCOL / 2);          // 0..191
    int cp  = idx - k * (gck::NCOL / 2);      // 0..64 (consecutive tid -> coalesced)
    int col = cp * 2;
    int c = k & 63;
    int r = k >> 6;
    const float2 v = *(const float2*)(x + ((size_t)c * gck::Hin + (y + r)) * gck::Win + x0 + col);
    xpanel[(col    ) * gck::KPAD + k] = (__bf16)v.x;
    xpanel[(col + 1) * gck::KPAD + k] = (__bf16)v.y;
  }

  if (tid < 32) __builtin_amdgcn_s_wait_tensorcnt(0);
  __syncthreads();

  const int lane = tid & 31;
  const int wave = tid >> 5;
  const int pix0 = wave * 16;
  const int n    = lane & 15;
  const int hi   = lane >> 4;

  v8f acc0 = {}, acc1 = {}, acc2 = {}, acc3 = {};
  const v16bf* __restrict__ Ald = (const v16bf*)apanel;

#pragma unroll
  for (int s = 0; s < 3; ++s) {
    const int col = pix0 + n + s;             // <= 129
#pragma unroll
    for (int kc = 0; kc < 6; ++kc) {
      v16bf b = *(const v16bf*)&xpanel[col * gck::KPAD + kc * 32 + hi * 16];
      int f = ((s * 6 + kc) * 4) * 32 + lane;
      v16bf a0 = Ald[f];
      v16bf a1 = Ald[f + 32];
      v16bf a2 = Ald[f + 64];
      v16bf a3 = Ald[f + 96];
      acc0 = __builtin_amdgcn_wmma_f32_16x16x32_bf16(false, a0, false, b,
                                                     (short)0, acc0, false, false);
      acc1 = __builtin_amdgcn_wmma_f32_16x16x32_bf16(false, a1, false, b,
                                                     (short)0, acc1, false, false);
      acc2 = __builtin_amdgcn_wmma_f32_16x16x32_bf16(false, a2, false, b,
                                                     (short)0, acc2, false, false);
      acc3 = __builtin_amdgcn_wmma_f32_16x16x32_bf16(false, a3, false, b,
                                                     (short)0, acc3, false, false);
    }
  }

  // ---- store: C/D layout: vgpr j, lane L -> M = j + 8*(L/16), N = L%16 ----
  const int px = x0 + pix0 + n;
  float* __restrict__ op = out + (size_t)y * gck::Wo + px;
  const size_t plane = (size_t)gck::Ho * gck::Wo;
#pragma unroll
  for (int j = 0; j < 8; ++j) {
    int mrow = j + 8 * hi;
    op[(size_t)(mrow +  0) * plane] = acc0[j];
    op[(size_t)(mrow + 16) * plane] = acc1[j];
    op[(size_t)(mrow + 32) * plane] = acc2[j];
    op[(size_t)(mrow + 48) * plane] = acc3[j];
  }
}

extern "C" void kernel_launch(void* const* d_in, const int* in_sizes, int n_in,
                              void* d_out, int out_size, void* d_ws, size_t ws_size,
                              hipStream_t stream) {
  const float* x        = (const float*)d_in[0];   // (1,64,514,514) f32
  const float* linCombs = (const float*)d_in[1];   // (64,576) f32
  float* out            = (float*)d_out;           // (1,64,512,512) f32
  __bf16* apack         = (__bf16*)d_ws;           // 73728 B of packed A fragments

  // Fold linCombs -> effective bf16 kernels in A-fragment layout (36864 elems).
  gck_fold_kernels<<<gck::APACK_ELTS / 256, 256, 0, stream>>>(linCombs, apack);

  // 512 rows x 4 pixel-tiles = 2048 blocks.
  gck_conv_wmma<<<(gck::Ho * gck::Wo) / gck::TILE_P, 256, 0, stream>>>(x, apack, out);
}
